// SessionGraph4_66073776881731
// MI455X (gfx1250) — matrix operand order repaired and checked by reading, not compiled, not run on hardware
//
#include <hip/hip_runtime.h>
#include <hip/hip_bf16.h>
#include <math.h>

#define DIM 128
#define H 8
#define NH 1024           // H*DIM
#define N_ITEMS 81920
#define E_INT 163840
#define N_T 4096
#define NUM_NODE 40000
#define NEG_SLOPE 0.2f
#define SCALE 12.0f

typedef __attribute__((ext_vector_type(16))) _Float16 v16h;
typedef __attribute__((ext_vector_type(8)))  _Float16 half8;
typedef __attribute__((ext_vector_type(8)))  float    v8f;

// ---------------------------------------------------------------- utilities

__global__ void fill_f32(float* __restrict__ p, float v, size_t n) {
    size_t i = (size_t)blockIdx.x * blockDim.x + threadIdx.x;
    size_t s = (size_t)gridDim.x * blockDim.x;
    for (; i < n; i += s) p[i] = v;
}

__global__ void cvt_f32_to_f16(const float* __restrict__ in, _Float16* __restrict__ out, size_t n) {
    size_t i = (size_t)blockIdx.x * blockDim.x + threadIdx.x;
    size_t s = (size_t)gridDim.x * blockDim.x;
    for (; i < n; i += s) out[i] = (_Float16)in[i];
}

__device__ __forceinline__ float block_reduce_sum_128(float v, float* red) {
    int t = threadIdx.x;
    red[t] = v;
    __syncthreads();
#pragma unroll
    for (int s = 64; s > 0; s >>= 1) {
        if (t < s) red[t] += red[t + s];
        __syncthreads();
    }
    float r = red[0];
    __syncthreads();
    return r;
}

__device__ __forceinline__ void atomicMaxF(float* addr, float val) {
    int* ai = (int*)addr;
    int old = __float_as_int(*addr);
    while (__int_as_float(old) < val) {
        int assumed = old;
        old = atomicCAS(ai, assumed, __float_as_int(val));
        if (old == assumed) break;
    }
}

// -------------------------------------------------- normalize / gather rows

// out_f16[n] = normalize(in[n]). 128 threads/block, 1 block/row.
__global__ void norm_rows_f16(const float* __restrict__ in, _Float16* __restrict__ out) {
    __shared__ float red[DIM];
    int n = blockIdx.x, d = threadIdx.x;
    float v = in[(size_t)n * DIM + d];
    float ss = block_reduce_sum_128(v * v, red);
    float nm = fmaxf(sqrtf(ss), 1e-12f);
    out[(size_t)n * DIM + d] = (_Float16)(v / nm);
}

// h_v = normalize(emb[iid]); write f32 + f16.
__global__ void hv_kernel(const float* __restrict__ emb, const int* __restrict__ iid,
                          float* __restrict__ hv, _Float16* __restrict__ hvh) {
    __shared__ float red[DIM];
    int n = blockIdx.x, d = threadIdx.x;
    int id = iid[n];
    float v = emb[(size_t)id * DIM + d];
    float ss = block_reduce_sum_128(v * v, red);
    float nm = fmaxf(sqrtf(ss), 1e-12f);
    float o = v / nm;
    hv[(size_t)n * DIM + d] = o;
    hvh[(size_t)n * DIM + d] = (_Float16)o;
}

// h = normalize(h1 + h2)
__global__ void combine_norm(const float* __restrict__ h1, const float* __restrict__ h2,
                             float* __restrict__ h) {
    __shared__ float red[DIM];
    int n = blockIdx.x, d = threadIdx.x;
    float v = h1[(size_t)n * DIM + d] + h2[(size_t)n * DIM + d];
    float ss = block_reduce_sum_128(v * v, red);
    float nm = fmaxf(sqrtf(ss), 1e-12f);
    h[(size_t)n * DIM + d] = v / nm;
}

// ---------------------------------------------------------- WMMA f16 GEMM
// C[M,N] = A[M,K] * Brow[N,K]^T  (X @ W.T with W row-major [N,K]).
// One wave -> one 16x64 macro-tile (4 accumulators); the A fragment is
// loaded once per K-step and reused against 4 B fragments, quadrupling
// WMMA density per byte of fragment traffic. K compile-time, mult of 32.
template <int KK, bool OUT_F16>
__global__ void wmma_gemm(const _Float16* __restrict__ A, const _Float16* __restrict__ Brow,
                          float* __restrict__ Cf, _Float16* __restrict__ Ch, int M, int N) {
    int wave = (int)((blockIdx.x * blockDim.x + threadIdx.x) >> 5);
    int lane = threadIdx.x & 31;
    int ntiles = N >> 6;                 // 64-wide macro tiles
    int mt = wave / ntiles;
    int nt = wave - mt * ntiles;
    if (mt >= (M >> 4)) return;          // wave-uniform: EXEC all-ones inside
    int g = lane >> 4;                   // half-wave group
    int r = lane & 15;
    const _Float16* arow = A + (size_t)(mt * 16 + r) * KK;
    const _Float16* brow0 = Brow + (size_t)(nt * 64 + r) * KK;
    v8f c0 = {}, c1 = {}, c2 = {}, c3 = {};
#pragma unroll
    for (int k0 = 0; k0 < KK; k0 += 32) {
        v16h a = *(const v16h*)(arow + k0 + 16 * g);
        v16h b0 = *(const v16h*)(brow0 + (size_t)(0 * 16) * KK + k0 + 16 * g);
        v16h b1 = *(const v16h*)(brow0 + (size_t)(1 * 16) * KK + k0 + 16 * g);
        v16h b2 = *(const v16h*)(brow0 + (size_t)(2 * 16) * KK + k0 + 16 * g);
        v16h b3 = *(const v16h*)(brow0 + (size_t)(3 * 16) * KK + k0 + 16 * g);
        c0 = __builtin_amdgcn_wmma_f32_16x16x32_f16(false, a, false, b0, (short)0, c0, false, false);
        c1 = __builtin_amdgcn_wmma_f32_16x16x32_f16(false, a, false, b1, (short)0, c1, false, false);
        c2 = __builtin_amdgcn_wmma_f32_16x16x32_f16(false, a, false, b2, (short)0, c2, false, false);
        c3 = __builtin_amdgcn_wmma_f32_16x16x32_f16(false, a, false, b3, (short)0, c3, false, false);
    }
#pragma unroll
    for (int j = 0; j < 4; ++j) {
        v8f c = (j == 0) ? c0 : (j == 1) ? c1 : (j == 2) ? c2 : c3;
        int col = nt * 64 + j * 16 + r;
#pragma unroll
        for (int rr = 0; rr < 8; ++rr) {
            int row = mt * 16 + rr + 8 * g;
            if (OUT_F16) Ch[(size_t)row * N + col] = (_Float16)c[rr];
            else         Cf[(size_t)row * N + col] = c[rr];
        }
    }
}

// ------------------------------------------------------------- GAT kernels

// el[n,h] = dot(fs[n,h,:], al[h,:]); er likewise. 1 block/node, 128 thr.
__global__ void gat_el_er(const _Float16* __restrict__ fs, const float* __restrict__ al,
                          const float* __restrict__ ar, float* __restrict__ el,
                          float* __restrict__ er) {
    __shared__ float red[DIM];
    int n = blockIdx.x, t = threadIdx.x;
#pragma unroll
    for (int h = 0; h < H; ++h) {
        float f = (float)fs[(size_t)n * NH + h * DIM + t];
        float sl = block_reduce_sum_128(f * al[h * DIM + t], red);
        if (t == 0) el[n * H + h] = sl;
        float sr = block_reduce_sum_128(f * ar[h * DIM + t], red);
        if (t == 0) er[n * H + h] = sr;
    }
}

// e = leaky_relu(el[src]+er[dst]); store; atomic segment-max into emax[dst].
__global__ void gat_edge_max(const int* __restrict__ src, const int* __restrict__ dst,
                             const float* __restrict__ el, const float* __restrict__ er,
                             float* __restrict__ ebuf, float* __restrict__ emax) {
    int i = blockIdx.x * blockDim.x + threadIdx.x;
    if (i >= E_INT * H) return;
    int e = i >> 3, h = i & 7;
    int s = src[e], d = dst[e];
    float v = el[s * H + h] + er[d * H + h];
    v = (v >= 0.f) ? v : NEG_SLOPE * v;
    ebuf[i] = v;
    atomicMaxF(&emax[d * H + h], v);
}

// ex = exp(e - emax[dst]); store; atomic segment-sum into denom[dst].
__global__ void gat_edge_exp(const int* __restrict__ dst, float* __restrict__ ebuf,
                             const float* __restrict__ emax, float* __restrict__ denom) {
    int i = blockIdx.x * blockDim.x + threadIdx.x;
    if (i >= E_INT * H) return;
    int e = i >> 3, h = i & 7;
    int d = dst[e];
    float ex = expf(ebuf[i] - emax[d * H + h]);
    ebuf[i] = ex;
    atomicAdd(&denom[d * H + h], ex);
}

// rst[dst] += fs[src] * a  (a = ex/denom per head). 1 block/edge, 256 thr.
__global__ void gat_edge_agg(const int* __restrict__ src, const int* __restrict__ dst,
                             const float* __restrict__ ebuf, const float* __restrict__ denom,
                             const _Float16* __restrict__ fs, float* __restrict__ rst) {
    __shared__ float a[H];
    int e = blockIdx.x;
    int d = dst[e];
    if (threadIdx.x < H)
        a[threadIdx.x] = ebuf[e * H + threadIdx.x] / denom[d * H + threadIdx.x];
    __syncthreads();
    int s = src[e];
#pragma unroll
    for (int t = threadIdx.x; t < NH; t += 256) {
        int h = t >> 7;
        atomicAdd(&rst[(size_t)d * NH + t], (float)fs[(size_t)s * NH + t] * a[h]);
    }
}

// out[n,d] = max_h(rst[n,h,d] + bias[h,d]) + feat[n,d]
__global__ void gat_finalize(const float* __restrict__ rst, const float* __restrict__ feat,
                             const float* __restrict__ bias, float* __restrict__ out) {
    int n = blockIdx.x, d = threadIdx.x;
    float m = -3.4e38f;
#pragma unroll
    for (int h = 0; h < H; ++h)
        m = fmaxf(m, rst[(size_t)n * NH + h * DIM + d] + bias[h * DIM + d]);
    out[(size_t)n * DIM + d] = m + feat[(size_t)n * DIM + d];
}

// ------------------------------------------------------------ aggregator

// per agg-edge: w = sum_d tanh(cat(ft,hp) . q_w[d]) * h_t[agg_dst][d];
// sr_g[agg_dst] += ft * w.   1 block/edge, 128 threads.
__global__ void pos_agg(const float* __restrict__ h, const int* __restrict__ agg_src,
                        const int* __restrict__ agg_dst, const int* __restrict__ pid,
                        const int* __restrict__ tid, const float* __restrict__ pos_emb,
                        const float* __restrict__ target_emb, const float* __restrict__ qw,
                        float* __restrict__ sr_g) {
    __shared__ float ft[DIM], hp[DIM], red[DIM];
    int e = blockIdx.x, t = threadIdx.x;
    int s = agg_src[e], d = agg_dst[e];
    ft[t] = h[(size_t)s * DIM + t];
    hp[t] = pos_emb[(size_t)pid[e] * DIM + t];
    __syncthreads();
    float acc = 0.f;
#pragma unroll 4
    for (int k = 0; k < DIM; ++k) acc += ft[k] * qw[t * 2 * DIM + k];
#pragma unroll 4
    for (int k = 0; k < DIM; ++k) acc += hp[k] * qw[t * 2 * DIM + DIM + k];
    float ht = target_emb[(size_t)tid[d] * DIM + t];
    float w = block_reduce_sum_128(tanhf(acc) * ht, red);
    atomicAdd(&sr_g[(size_t)d * DIM + t], ft[t] * w);
}

// sr[t] = cat(sr_g[t], h[last_nodes[t]]) @ fc_sr_w.T ; nsr = 12*normalize(sr) in f16
__global__ void sr_kernel(const float* __restrict__ sr_g, const float* __restrict__ h,
                          const int* __restrict__ last_nodes, const float* __restrict__ fw,
                          _Float16* __restrict__ nsr) {
    __shared__ float red[DIM];
    int t = blockIdx.x, d = threadIdx.x;
    int ln = last_nodes[t];
    float acc = 0.f;
#pragma unroll 4
    for (int k = 0; k < DIM; ++k) acc += sr_g[(size_t)t * DIM + k] * fw[d * 2 * DIM + k];
#pragma unroll 4
    for (int k = 0; k < DIM; ++k) acc += h[(size_t)ln * DIM + k] * fw[d * 2 * DIM + DIM + k];
    float ss = block_reduce_sum_128(acc * acc, red);
    float nm = fmaxf(sqrtf(ss), 1e-12f);
    nsr[(size_t)t * DIM + d] = (_Float16)(SCALE * acc / nm);
}

// ------------------------------------------------------------------ driver

static void run_gat_layer(hipStream_t stream, const float* hv, const _Float16* hvh,
                          const _Float16* fcw16, const float* al, const float* ar,
                          const float* bias, const int* src, const int* dst,
                          _Float16* fs16, float* el, float* er, float* ebuf,
                          float* emax, float* denom, float* rst, float* out) {
    // init segment buffers
    fill_f32<<<2048, 256, 0, stream>>>(emax, -3.4e38f, (size_t)N_ITEMS * H);
    fill_f32<<<2048, 256, 0, stream>>>(denom, 0.f, (size_t)N_ITEMS * H);
    fill_f32<<<4096, 256, 0, stream>>>(rst, 0.f, (size_t)N_ITEMS * NH);
    // fs = h_v @ fc_w.T  (WMMA, 16x64 macro-tiles)
    {
        int waves = (N_ITEMS / 16) * (NH / 64);
        wmma_gemm<DIM, true><<<waves / 8, 256, 0, stream>>>(hvh, fcw16, nullptr, fs16,
                                                            N_ITEMS, NH);
    }
    gat_el_er<<<N_ITEMS, DIM, 0, stream>>>(fs16, al, ar, el, er);
    int ethreads = E_INT * H;
    gat_edge_max<<<(ethreads + 255) / 256, 256, 0, stream>>>(src, dst, el, er, ebuf, emax);
    gat_edge_exp<<<(ethreads + 255) / 256, 256, 0, stream>>>(dst, ebuf, emax, denom);
    gat_edge_agg<<<E_INT, 256, 0, stream>>>(src, dst, ebuf, denom, fs16, rst);
    gat_finalize<<<N_ITEMS, DIM, 0, stream>>>(rst, hv, bias, out);
}

extern "C" void kernel_launch(void* const* d_in, const int* in_sizes, int n_in,
                              void* d_out, int out_size, void* d_ws, size_t ws_size,
                              hipStream_t stream) {
    (void)in_sizes; (void)n_in; (void)out_size; (void)ws_size;
    const int*   iid        = (const int*)d_in[0];
    const int*   tid        = (const int*)d_in[1];
    const int*   pid        = (const int*)d_in[2];
    const int*   src_i      = (const int*)d_in[3];
    const int*   dst_i      = (const int*)d_in[4];
    const int*   agg_src    = (const int*)d_in[5];
    const int*   agg_dst    = (const int*)d_in[6];
    const int*   last_nodes = (const int*)d_in[7];
    const float* emb        = (const float*)d_in[8];
    const float* pos_emb    = (const float*)d_in[9];
    const float* target_emb = (const float*)d_in[10];
    const float* gat1_fc    = (const float*)d_in[11];
    const float* gat1_al    = (const float*)d_in[12];
    const float* gat1_ar    = (const float*)d_in[13];
    const float* gat1_b     = (const float*)d_in[14];
    const float* gat2_fc    = (const float*)d_in[15];
    const float* gat2_al    = (const float*)d_in[16];
    const float* gat2_ar    = (const float*)d_in[17];
    const float* gat2_b     = (const float*)d_in[18];
    const float* q_w        = (const float*)d_in[19];
    const float* fc_sr_w    = (const float*)d_in[20];
    float* scores = (float*)d_out;

    // ---- carve workspace
    char* w = (char*)d_ws;
    auto carve = [&](size_t bytes) -> void* {
        void* p = (void*)w;
        w += (bytes + 255) & ~(size_t)255;
        return p;
    };
    float*    hv      = (float*)   carve((size_t)N_ITEMS * DIM * 4);
    _Float16* hvh     = (_Float16*)carve((size_t)N_ITEMS * DIM * 2);
    _Float16* nemb16  = (_Float16*)carve((size_t)NUM_NODE * DIM * 2);
    _Float16* fcw1    = (_Float16*)carve((size_t)NH * DIM * 2);
    _Float16* fcw2    = (_Float16*)carve((size_t)NH * DIM * 2);
    _Float16* fs16    = (_Float16*)carve((size_t)N_ITEMS * NH * 2);
    float*    el      = (float*)   carve((size_t)N_ITEMS * H * 4);
    float*    er      = (float*)   carve((size_t)N_ITEMS * H * 4);
    float*    ebuf    = (float*)   carve((size_t)E_INT * H * 4);
    float*    emax    = (float*)   carve((size_t)N_ITEMS * H * 4);
    float*    denom   = (float*)   carve((size_t)N_ITEMS * H * 4);
    float*    rst     = (float*)   carve((size_t)N_ITEMS * NH * 4);
    float*    h1      = (float*)   carve((size_t)N_ITEMS * DIM * 4);
    float*    h2      = (float*)   carve((size_t)N_ITEMS * DIM * 4);
    float*    hbuf    = (float*)   carve((size_t)N_ITEMS * DIM * 4);
    float*    sr_g    = (float*)   carve((size_t)N_T * DIM * 4);
    _Float16* nsr     = (_Float16*)carve((size_t)N_T * DIM * 2);

    // ---- precompute normalized embeddings + gathered node features
    norm_rows_f16<<<NUM_NODE, DIM, 0, stream>>>(emb, nemb16);
    hv_kernel<<<N_ITEMS, DIM, 0, stream>>>(emb, iid, hv, hvh);
    cvt_f32_to_f16<<<512, 256, 0, stream>>>(gat1_fc, fcw1, (size_t)NH * DIM);
    cvt_f32_to_f16<<<512, 256, 0, stream>>>(gat2_fc, fcw2, (size_t)NH * DIM);

    // ---- GAT layers (layer 2 runs on reversed edges)
    run_gat_layer(stream, hv, hvh, fcw1, gat1_al, gat1_ar, gat1_b,
                  src_i, dst_i, fs16, el, er, ebuf, emax, denom, rst, h1);
    run_gat_layer(stream, hv, hvh, fcw2, gat2_al, gat2_ar, gat2_b,
                  dst_i, src_i, fs16, el, er, ebuf, emax, denom, rst, h2);
    combine_norm<<<N_ITEMS, DIM, 0, stream>>>(h1, h2, hbuf);

    // ---- positional attention aggregator
    fill_f32<<<2048, 256, 0, stream>>>(sr_g, 0.f, (size_t)N_T * DIM);
    pos_agg<<<N_ITEMS, DIM, 0, stream>>>(hbuf, agg_src, agg_dst, pid, tid,
                                         pos_emb, target_emb, q_w, sr_g);
    sr_kernel<<<N_T, DIM, 0, stream>>>(sr_g, hbuf, last_nodes, fc_sr_w, nsr);

    // ---- scores = (12 * normalize(sr)) @ normalize(emb).T  (WMMA, 16x64 tiles)
    {
        int waves = (N_T / 16) * (NUM_NODE / 64);
        wmma_gemm<DIM, false><<<waves / 8, 256, 0, stream>>>(nsr, nemb16, scores, nullptr,
                                                             N_T, NUM_NODE);
    }
}
